// Encoder_87445534146869
// MI455X (gfx1250) — matrix operand order repaired and checked by reading, not compiled, not run on hardware
//
#include <hip/hip_runtime.h>
#include <hip/hip_bf16.h>

typedef __attribute__((ext_vector_type(16))) __bf16 v16bf;
typedef __attribute__((ext_vector_type(8)))  __bf16 v8bf;
typedef __attribute__((ext_vector_type(8)))  float  v8f;

#define BM 128
#define BN 128
#define BK 32
#define LDT 40   // padded LDS row stride in bf16 elems (80B: 16B-aligned b128 frags, bank-spread)

// ------------------------------------------------------------------
// Weight prep: OIHW fp32 (3x3) + [Cout][Ch] fp32  ->  bf16 [Cout][Ktot]
// K order: X part = (kh*3+kw)*Cin + ci  (9 segments of Cin), then H part (Ch)
// ------------------------------------------------------------------
__global__ void prep_w_kernel(const float* __restrict__ wih, const float* __restrict__ whh,
                              __bf16* __restrict__ out, int Cin, int cinShift,
                              int Khi, int Ch, int Ktot, long total) {
  long idx = (long)blockIdx.x * blockDim.x + threadIdx.x;
  if (idx >= total) return;
  int  k  = (int)(idx % Ktot);
  long co = idx / Ktot;
  float v;
  if (k < Khi) {
    int seg = k >> cinShift;        // kh*3+kw
    int ci  = k & (Cin - 1);
    v = wih[((size_t)co * Cin + ci) * 9 + seg];
  } else {
    v = whh[(size_t)co * Ch + (k - Khi)];
  }
  out[idx] = (__bf16)v;
}

// ------------------------------------------------------------------
// NCHW fp32 -> NHWC bf16 (for previous hidden states)
// ------------------------------------------------------------------
__global__ void to_nhwc_bf16(const float* __restrict__ src, __bf16* __restrict__ dst,
                             int C, int HW, long total) {
  long idx = (long)blockIdx.x * blockDim.x + threadIdx.x;   // over NHWC dst
  if (idx >= total) return;
  int  c    = (int)(idx % C);
  long rest = idx / C;
  int  sp   = (int)(rest % HW);
  long b    = rest / HW;
  dst[idx] = (__bf16)src[((b * C) + c) * (long)HW + sp];
}

// ------------------------------------------------------------------
// Stem: [16,3,256,256] -conv3x3 s2 p1-> NHWC bf16 [16,128,128,64]
// channel-fastest threads: coalesced NHWC stores, x reads broadcast
// ------------------------------------------------------------------
__global__ void stem_kernel(const float* __restrict__ x, const float* __restrict__ w,
                            __bf16* __restrict__ out) {
  int idx = blockIdx.x * blockDim.x + threadIdx.x;   // 16*128*128*64 threads
  int co = idx & 63;
  int ox = (idx >> 6) & 127;
  int oy = (idx >> 13) & 127;
  int b  = idx >> 20;
  float acc = 0.f;
#pragma unroll
  for (int ci = 0; ci < 3; ++ci)
#pragma unroll
    for (int kh = 0; kh < 3; ++kh) {
      int iy = oy * 2 + kh - 1;
      if (iy < 0 || iy > 255) continue;
#pragma unroll
      for (int kw = 0; kw < 3; ++kw) {
        int ix = ox * 2 + kw - 1;
        if (ix < 0 || ix > 255) continue;
        acc += x[((b * 3 + ci) * 256 + iy) * 256 + ix] *
               w[((co * 3 + ci) * 3 + kh) * 3 + kw];
      }
    }
  out[idx] = (__bf16)acc;   // idx == ((b*128+oy)*128+ox)*64+co
}

// ------------------------------------------------------------------
// Fused implicit-GEMM:  gates[Cout x N] = [Wih|Whh](bf16) * [im2col(X); H]
//   NHWC activations, (kh,kw)-major K -> B-stage is pure b128 traffic.
//   Double-buffered LDS, 8 waves, wave tile 64x32 (4x2 WMMA accumulators).
// ------------------------------------------------------------------
__global__ __launch_bounds__(256)
void gemm_conv_wmma(const __bf16* __restrict__ Wcat, const __bf16* __restrict__ X,
                    const __bf16* __restrict__ Hbf, float* __restrict__ gates,
                    int Cout, int Cin, int cinShift, int IH, int IW,
                    int OW, int owShift, int ohwShift,
                    int Khi, int Ch, int Ktot) {
  __shared__ __align__(16) __bf16 sA[2][BM * LDT];
  __shared__ __align__(16) __bf16 sB[2][BN * LDT];

  const int tid  = threadIdx.x;
  const int lane = tid & 31;
  const int wave = tid >> 5;
  const int wm   = wave & 1;    // 0..1 -> 64-row slab
  const int wn   = wave >> 1;   // 0..3 -> 32-col slab

  const int n0 = blockIdx.x * BN;
  const int m0 = blockIdx.y * BM;
  const int OHW = 1 << ohwShift;

  // B-loader coords: one pixel per thread, 16 consecutive channels (khalf half)
  const int bn    = tid & 127;
  const int khalf = tid >> 7;            // 0/1 -> k offset 16*khalf
  const int n_g = n0 + bn;
  const int bb  = n_g >> ohwShift;
  const int s   = n_g & (OHW - 1);
  const int oy  = s >> owShift;
  const int ox  = s & (OW - 1);

  // A-loader coords: 4 bf16 per chunk, 4 row-passes
  const int ar = tid >> 3;               // 0..31
  const int ak = (tid & 7) * 4;          // 0..28

  uint2 aR[4];
  v8bf  bR0, bR1;

  auto loadA = [&](int k0) {
#pragma unroll
    for (int p = 0; p < 4; ++p) {
      int row = ar + 32 * p;
      aR[p] = *(const uint2*)(Wcat + (size_t)(m0 + row) * Ktot + (k0 + ak));
    }
  };
  auto storeA = [&](int buf) {
#pragma unroll
    for (int p = 0; p < 4; ++p)
      *(uint2*)(&sA[buf][(ar + 32 * p) * LDT + ak]) = aR[p];
  };
  auto loadB = [&](int k0) {
    int kg = k0 + khalf * 16;
    const __bf16* src;
    bool valid = true;
    if (kg < Khi) {                       // X part (3x3, stride 2, pad 1)
      int seg  = kg >> cinShift;          // kh*3+kw
      int coff = kg & (Cin - 1);
      int kh = seg / 3;
      int kw = seg - kh * 3;
      int iy = 2 * oy + kh - 1;
      int ix = 2 * ox + kw - 1;
      valid = ((unsigned)iy < (unsigned)IH) & ((unsigned)ix < (unsigned)IW);
      src = X + (((size_t)(bb * IH + iy) * IW + ix) << cinShift) + coff;
    } else {                              // H part (1x1), NHWC
      int coff = kg - Khi;
      src = Hbf + (size_t)((bb << ohwShift) + s) * Ch + coff;
    }
    if (valid) {
      bR0 = *(const v8bf*)src;
      bR1 = *(const v8bf*)(src + 8);
    } else {
      const __bf16 z = (__bf16)0.f;
      bR0 = v8bf{z, z, z, z, z, z, z, z};
      bR1 = bR0;
    }
  };
  auto storeB = [&](int buf) {
    *(v8bf*)(&sB[buf][bn * LDT + khalf * 16])     = bR0;
    *(v8bf*)(&sB[buf][bn * LDT + khalf * 16 + 8]) = bR1;
  };

  v8f acc[4][2];
#pragma unroll
  for (int tm = 0; tm < 4; ++tm)
#pragma unroll
    for (int tn = 0; tn < 2; ++tn)
      acc[tm][tn] = v8f{0.f, 0.f, 0.f, 0.f, 0.f, 0.f, 0.f, 0.f};

  // prologue: stage tile 0
  loadA(0); loadB(0);
  storeA(0); storeB(0);

  const int nT = Ktot / BK;
  const int lm   = lane & 15;
  const int aoff = (lane >> 4) * 16;     // A frag K halves
  const int boff = (lane >> 4) * 16;     // B frag K halves

  for (int t = 0; t < nT; ++t) {
    __syncthreads();
    const int cur = t & 1;
    if (t + 1 < nT) { loadA((t + 1) * BK); loadB((t + 1) * BK); }

    v16bf afrag[4];
#pragma unroll
    for (int tm = 0; tm < 4; ++tm) {
      int row = wm * 64 + tm * 16 + lm;
      union { v16bf v; v8bf h[2]; } u;
      u.h[0] = *(const v8bf*)&sA[cur][row * LDT + aoff];
      u.h[1] = *(const v8bf*)&sA[cur][row * LDT + aoff + 8];
      afrag[tm] = u.v;
    }
    v16bf bfrag[2];
#pragma unroll
    for (int tn = 0; tn < 2; ++tn) {
      int nn = wn * 32 + tn * 16 + lm;
      union { v16bf v; v8bf h[2]; } u;
      u.h[0] = *(const v8bf*)&sB[cur][nn * LDT + boff];
      u.h[1] = *(const v8bf*)&sB[cur][nn * LDT + boff + 8];
      bfrag[tn] = u.v;
    }

#pragma unroll
    for (int tm = 0; tm < 4; ++tm)
#pragma unroll
      for (int tn = 0; tn < 2; ++tn)
        acc[tm][tn] = __builtin_amdgcn_wmma_f32_16x16x32_bf16(
            false, afrag[tm], false, bfrag[tn], (short)0, acc[tm][tn],
            false, false);

    if (t + 1 < nT) { storeA(cur ^ 1); storeB(cur ^ 1); }
  }

  // ---- store gates (NCHW fp32): lane<16: M=i, lane>=16: M=8+i, N=lane%16 ----
  const int nloc = lane & 15;
  const int mhi  = (lane >> 4) * 8;
#pragma unroll
  for (int tm = 0; tm < 4; ++tm)
#pragma unroll
    for (int tn = 0; tn < 2; ++tn) {
      int co_base = m0 + wm * 64 + tm * 16 + mhi;
      int n  = n0 + wn * 32 + tn * 16 + nloc;
      int b2 = n >> ohwShift;
      int s2 = n & (OHW - 1);
      float* dst = gates + ((size_t)b2 * Cout + co_base) * OHW + s2;
#pragma unroll
      for (int i = 0; i < 8; ++i)
        dst[(size_t)i * OHW] = acc[tm][tn][i];
    }
}

// ------------------------------------------------------------------
// LSTM pointwise: gates [B,4C,HW] (i,f,g,o) + c_prev -> h,c fp32 (NCHW)
// plus next-layer h in NHWC bf16
// ------------------------------------------------------------------
__global__ void lstm_kernel(const float* __restrict__ gates, const float* __restrict__ c_prev,
                            float* __restrict__ h_out, float* __restrict__ h_out2,
                            float* __restrict__ c_out, __bf16* __restrict__ h_bf,
                            int C, int HW, long total) {
  long idx = (long)blockIdx.x * blockDim.x + threadIdx.x;
  if (idx >= total) return;
  int  sp = (int)(idx % HW);
  int  ch = (int)((idx / HW) % C);
  long b  = idx / ((long)HW * C);
  long base = (b * 4 * C + ch) * (long)HW + sp;
  long gstride = (long)C * HW;
  float ig = gates[base];
  float fg = gates[base + gstride];
  float gg = gates[base + 2 * gstride];
  float og = gates[base + 3 * gstride];
  float si = 1.f / (1.f + expf(-ig));
  float sf = 1.f / (1.f + expf(-fg));
  float so = 1.f / (1.f + expf(-og));
  float cn = sf * c_prev[idx] + si * tanhf(gg);
  float hn = so * tanhf(cn);
  h_out[idx] = hn;
  c_out[idx] = cn;
  if (h_out2) h_out2[idx] = hn;
  if (h_bf)   h_bf[(b * HW + sp) * (long)C + ch] = (__bf16)hn;   // NHWC
}

// ------------------------------------------------------------------
extern "C" void kernel_launch(void* const* d_in, const int* in_sizes, int n_in,
                              void* d_out, int out_size, void* d_ws, size_t ws_size,
                              hipStream_t stream) {
  const float* x     = (const float*)d_in[0];
  const float* h1    = (const float*)d_in[1];
  const float* c1    = (const float*)d_in[2];
  const float* h2    = (const float*)d_in[3];
  const float* c2    = (const float*)d_in[4];
  const float* h3    = (const float*)d_in[5];
  const float* c3    = (const float*)d_in[6];
  const float* wconv = (const float*)d_in[7];
  const float* wih1  = (const float*)d_in[8];
  const float* whh1  = (const float*)d_in[9];
  const float* wih2  = (const float*)d_in[10];
  const float* whh2  = (const float*)d_in[11];
  const float* wih3  = (const float*)d_in[12];
  const float* whh3  = (const float*)d_in[13];
  float* out = (float*)d_out;

  // ---- workspace carve-up (~425 MB; gates buffer reused across layers) ----
  char* ws = (char*)d_ws;
  size_t off = 0;
  auto take = [&](size_t bytes) -> char* {
    char* p = ws + off;
    off = (off + bytes + 255) & ~(size_t)255;
    return p;
  };
  __bf16* wcat1   = (__bf16*)take((size_t)1024 * 832 * 2);
  __bf16* wcat2   = (__bf16*)take((size_t)2048 * 2816 * 2);
  __bf16* wcat3   = (__bf16*)take((size_t)2048 * 5120 * 2);
  __bf16* stem_bf = (__bf16*)take((size_t)16 * 64 * 128 * 128 * 2);   // NHWC
  __bf16* h1n_bf  = (__bf16*)take((size_t)16 * 256 * 64 * 64 * 2);    // NHWC
  __bf16* h2n_bf  = (__bf16*)take((size_t)16 * 512 * 32 * 32 * 2);    // NHWC
  __bf16* h1p_bf  = (__bf16*)take((size_t)16 * 256 * 64 * 64 * 2);    // NHWC prev-h
  __bf16* h2p_bf  = (__bf16*)take((size_t)16 * 512 * 32 * 32 * 2);
  __bf16* h3p_bf  = (__bf16*)take((size_t)16 * 512 * 16 * 16 * 2);
  float*  gates   = (float*) take((size_t)16 * 1024 * 64 * 64 * 4);   // max layer

  // d_out element offsets: (h3n, h1n, c1n, h2n, c2n, h3n, c3n)
  const size_t OFF_H3A = 0;
  const size_t OFF_H1  = 2097152;
  const size_t OFF_C1  = 18874368;
  const size_t OFF_H2  = 35651584;
  const size_t OFF_C2  = 44040192;
  const size_t OFF_H3B = 52428800;
  const size_t OFF_C3  = 54525952;

  // ---- weight prep (fp32 -> bf16, (kh,kw)-major | hh concat) ----
  {
    long t1 = (long)1024 * 832, t2 = (long)2048 * 2816, t3 = (long)2048 * 5120;
    prep_w_kernel<<<dim3((unsigned)((t1 + 255) / 256)), 256, 0, stream>>>(wih1, whh1, wcat1, 64,  6, 576,  256, 832,  t1);
    prep_w_kernel<<<dim3((unsigned)((t2 + 255) / 256)), 256, 0, stream>>>(wih2, whh2, wcat2, 256, 8, 2304, 512, 2816, t2);
    prep_w_kernel<<<dim3((unsigned)((t3 + 255) / 256)), 256, 0, stream>>>(wih3, whh3, wcat3, 512, 9, 4608, 512, 5120, t3);
  }
  // ---- prev-h NCHW fp32 -> NHWC bf16 ----
  {
    long t1 = (long)16 * 256 * 4096, t2 = (long)16 * 512 * 1024, t3 = (long)16 * 512 * 256;
    to_nhwc_bf16<<<dim3((unsigned)((t1 + 255) / 256)), 256, 0, stream>>>(h1, h1p_bf, 256, 4096, t1);
    to_nhwc_bf16<<<dim3((unsigned)((t2 + 255) / 256)), 256, 0, stream>>>(h2, h2p_bf, 512, 1024, t2);
    to_nhwc_bf16<<<dim3((unsigned)((t3 + 255) / 256)), 256, 0, stream>>>(h3, h3p_bf, 512, 256,  t3);
  }

  // ---- stem conv (NHWC bf16 out) ----
  stem_kernel<<<dim3(16 * 64 * 128 * 128 / 256), 256, 0, stream>>>(x, wconv, stem_bf);

  // ---- layer 1: M=1024, N=65536, K=832 (IH=IW=128, OH=OW=64) ----
  gemm_conv_wmma<<<dim3(65536 / BN, 1024 / BM), 256, 0, stream>>>(
      wcat1, stem_bf, h1p_bf, gates, 1024, 64, 6, 128, 128, 64, 6, 12, 576, 256, 832);
  lstm_kernel<<<dim3((unsigned)(((long)16 * 256 * 4096 + 255) / 256)), 256, 0, stream>>>(
      gates, c1, out + OFF_H1, nullptr, out + OFF_C1, h1n_bf, 256, 4096, (long)16 * 256 * 4096);

  // ---- layer 2: M=2048, N=16384, K=2816 (IH=64, OH=32) ----
  gemm_conv_wmma<<<dim3(16384 / BN, 2048 / BM), 256, 0, stream>>>(
      wcat2, h1n_bf, h2p_bf, gates, 2048, 256, 8, 64, 64, 32, 5, 10, 2304, 512, 2816);
  lstm_kernel<<<dim3((unsigned)(((long)16 * 512 * 1024 + 255) / 256)), 256, 0, stream>>>(
      gates, c2, out + OFF_H2, nullptr, out + OFF_C2, h2n_bf, 512, 1024, (long)16 * 512 * 1024);

  // ---- layer 3: M=2048, N=4096, K=5120 (IH=32, OH=16) ----
  gemm_conv_wmma<<<dim3(4096 / BN, 2048 / BM), 256, 0, stream>>>(
      wcat3, h2n_bf, h3p_bf, gates, 2048, 512, 9, 32, 32, 16, 4, 8, 4608, 512, 5120);
  lstm_kernel<<<dim3((unsigned)(((long)16 * 512 * 256 + 255) / 256)), 256, 0, stream>>>(
      gates, c3, out + OFF_H3A, out + OFF_H3B, out + OFF_C3, nullptr, 512, 256, (long)16 * 512 * 256);
}
